// equivariant_encoder_5566277616514
// MI455X (gfx1250) — compile-verified, matrix operand order checked by reading, not run on hardware
//
#include <hip/hip_runtime.h>
#include <math.h>

typedef __attribute__((ext_vector_type(16))) _Float16 v16h;
typedef __attribute__((ext_vector_type(8)))  float    v8f;

#define NNODES 200000
#define GG     8192
#define MOUT   85

// K index held by element e of a 16-element f16 fragment register in `lane`,
// per the documented 16-bit A-matrix 16x32 layout (B assumed mirrored).
__device__ __forceinline__ int kmap(int lane, int e) {
    return ((e >> 3) << 4) + ((lane >> 4) << 3) + (e & 7);
}

// tanh-approx gelu, rewritten as a sigmoid and evaluated branch-free with the
// hardware v_exp_f32 / v_rcp_f32: 0.5*x*(1+tanh(c*u)) == x * sigmoid(2c*u).
__device__ __forceinline__ float gelu_f(float x) {
    float u = x + 0.044715f * x * x * x;
    float t = __expf(-1.5957691216057308f * u);   // -2*sqrt(2/pi)*u
    return x * __builtin_amdgcn_rcpf(1.0f + t);
}

// Convert an f32 weight matrix (Kdim x 64, row major) into f16 WMMA B-fragments
// in LDS: layout [kt][nt][lane][e] (frag = 512 halfs).
__device__ void fill_wfrags(const float* __restrict__ W, int Kdim,
                            _Float16* dst, int tid, int nthreads) {
    int total = (Kdim / 32) * 4 * 32 * 16;
    for (int i = tid; i < total; i += nthreads) {
        int e    = i & 15;
        int lane = (i >> 4) & 31;
        int nt   = (i >> 9) & 3;
        int kt   = i >> 11;
        int k = kt * 32 + kmap(lane, e);
        int n = nt * 16 + (lane & 15);
        dst[i] = (_Float16)W[k * 64 + n];
    }
}

// Load one A fragment (16 x 32 k-tile) from per-wave LDS scratch (ld = 96 halfs).
__device__ __forceinline__ v16h load_afrag(const _Float16* s, int lane, int kt) {
    int m  = lane & 15;
    int kb = kt * 32 + ((lane >> 4) << 3);
    v16h a;
#pragma unroll
    for (int e = 0; e < 8; ++e) a[e]     = s[m * 96 + kb + e];
#pragma unroll
    for (int e = 0; e < 8; ++e) a[8 + e] = s[m * 96 + kb + 16 + e];
    return a;
}

// One 16x(KT*32) @ (KT*32)x64 layer: A from scratch, B frags from LDS,
// bias + optional gelu, f16 result written back to scratch cols 0..63.
template <int KT, bool GELU>
__device__ __forceinline__ void mlp_layer(const _Float16* __restrict__ wf,
                                          const float* __restrict__ bias,
                                          _Float16* scr, int lane) {
    v16h a[KT];
#pragma unroll
    for (int kt = 0; kt < KT; ++kt) a[kt] = load_afrag(scr, lane, kt);
    int n16    = lane & 15;
    int rowoff = (lane >> 4) * 8;
#pragma unroll
    for (int nt = 0; nt < 4; ++nt) {
        v8f c = {};
#pragma unroll
        for (int kt = 0; kt < KT; ++kt) {
            const v16h b = *(const v16h*)&wf[((kt * 4 + nt) * 32 + lane) * 16];
            c = __builtin_amdgcn_wmma_f32_16x16x32_f16(false, a[kt], false, b,
                                                       (short)0, c, false, false);
        }
        int n = nt * 16 + n16;
        float bv = bias[n];
#pragma unroll
        for (int r = 0; r < 8; ++r) {
            float v = c[r] + bv;
            if (GELU) v = gelu_f(v);
            scr[(rowoff + r) * 96 + n] = (_Float16)v;
        }
    }
}

#define WAVES 4

__global__ __launch_bounds__(128) void node_kernel(
    const float* __restrict__ pos, const int* __restrict__ z,
    const int* __restrict__ batch, const float* __restrict__ node_table,
    const float* __restrict__ W_rbf0, const float* __restrict__ b_rbf0,
    const float* __restrict__ W_rbf1, const float* __restrict__ b_rbf1,
    const float* __restrict__ W_m0, const float* __restrict__ b_m0,
    const float* __restrict__ W_m1, const float* __restrict__ b_m1,
    const float* __restrict__ W_m2, const float* __restrict__ b_m2,
    const float* __restrict__ w_tp1,
    float* __restrict__ cnt, float* __restrict__ s1, int Nn)
{
    __shared__ alignas(32) _Float16 wf_rbf0[1 * 4 * 512];
    __shared__ alignas(32) _Float16 wf_rbf1[2 * 4 * 512];
    __shared__ alignas(32) _Float16 wf_m0  [3 * 4 * 512];
    __shared__ alignas(32) _Float16 wf_m1  [2 * 4 * 512];
    __shared__ alignas(32) _Float16 wf_m2  [2 * 4 * 512];
    __shared__ alignas(32) _Float16 wf_tp1 [2 * 4 * 512];
    __shared__ alignas(32) _Float16 scratch[WAVES][16 * 96];
    __shared__ float shb[WAVES][16 * 3];
    __shared__ int   gbuf[WAVES][16];

    const int tid = threadIdx.x;
    fill_wfrags(W_rbf0, 32, wf_rbf0, tid, blockDim.x);
    fill_wfrags(W_rbf1, 64, wf_rbf1, tid, blockDim.x);
    fill_wfrags(W_m0,   96, wf_m0,   tid, blockDim.x);
    fill_wfrags(W_m1,   64, wf_m1,   tid, blockDim.x);
    fill_wfrags(W_m2,   64, wf_m2,   tid, blockDim.x);
    fill_wfrags(w_tp1,  64, wf_tp1,  tid, blockDim.x);
    __syncthreads();

    const int wave = tid >> 5;   // wave32
    const int lane = tid & 31;
    const int tile = blockIdx.x * WAVES + wave;
    const int base = tile * 16;
    if (base >= Nn) return;

    _Float16* scr = scratch[wave];
    const int m = lane & 15;
    const int node  = base + m;
    const bool valid = node < Nn;

    float px = 0.f, py = 0.f, pz = 0.f;
    if (valid) { px = pos[node*3+0]; py = pos[node*3+1]; pz = pos[node*3+2]; }
    const float dist = sqrtf(px*px + py*py + pz*pz);

    // radial basis -> A fragment (16 x 32, K = 32 exactly one k-tile)
    const float step  = 1.0f / 31.0f;
    const float coeff = -0.5f / (step * step);
    v16h arbf;
#pragma unroll
    for (int e = 0; e < 16; ++e) {
        int k = kmap(lane, e);
        float d = dist - (float)k * step;
        arbf[e] = valid ? (_Float16)__expf(coeff * d * d) : (_Float16)0.0f;
    }

    if (lane < 16) {
        const float s3i = 1.7320508075688772f * 0.125f;  // sqrt(3) * 1/sqrt(64)
        shb[wave][m*3+0] = s3i * px;
        shb[wave][m*3+1] = s3i * py;
        shb[wave][m*3+2] = s3i * pz;
        int g = -1;
        if (valid) { g = batch[node]; atomicAdd(&cnt[g], 1.0f); }
        gbuf[wave][m] = g;
        int zz = valid ? z[node] : 0;
#pragma unroll
        for (int e = 0; e < 32; ++e)
            scr[m * 96 + 64 + e] = valid ? (_Float16)node_table[zz * 32 + e]
                                         : (_Float16)0.0f;
    }

    // layer 1: r1 = gelu(rbf @ W_rbf0 + b) ; A from registers
    {
        const int n16 = lane & 15, rowoff = (lane >> 4) * 8;
#pragma unroll
        for (int nt = 0; nt < 4; ++nt) {
            v8f c = {};
            const v16h b = *(const v16h*)&wf_rbf0[(nt * 32 + lane) * 16];
            c = __builtin_amdgcn_wmma_f32_16x16x32_f16(false, arbf, false, b,
                                                       (short)0, c, false, false);
            int n = nt * 16 + n16;
            float bv = b_rbf0[n];
#pragma unroll
            for (int r = 0; r < 8; ++r)
                scr[(rowoff + r) * 96 + n] = (_Float16)gelu_f(c[r] + bv);
        }
    }
    mlp_layer<2, false>(wf_rbf1, b_rbf1, scr, lane);  // r = r1@W_rbf1 + b
    mlp_layer<3, true >(wf_m0,   b_m0,   scr, lane);  // h = gelu([r,table]@W_m0+b)
    mlp_layer<2, true >(wf_m1,   b_m1,   scr, lane);  // h = gelu(h@W_m1+b)
    mlp_layer<2, false>(wf_m2,   b_m2,   scr, lane);  // msg = h@W_m2+b

    // t1 = msg @ w_tp1 ; scatter t1 * sh1 * inv into s1[g][u][comp]
    {
        v16h a0 = load_afrag(scr, lane, 0);
        v16h a1 = load_afrag(scr, lane, 1);
        const int rowoff = (lane >> 4) * 8;
        float sx[8], sy[8], sz[8]; int gg[8];
#pragma unroll
        for (int r = 0; r < 8; ++r) {
            int rm = rowoff + r;
            sx[r] = shb[wave][rm*3+0];
            sy[r] = shb[wave][rm*3+1];
            sz[r] = shb[wave][rm*3+2];
            gg[r] = gbuf[wave][rm];
        }
#pragma unroll
        for (int nt = 0; nt < 4; ++nt) {
            v8f c = {};
            const v16h b0 = *(const v16h*)&wf_tp1[((0 * 4 + nt) * 32 + lane) * 16];
            const v16h b1 = *(const v16h*)&wf_tp1[((1 * 4 + nt) * 32 + lane) * 16];
            c = __builtin_amdgcn_wmma_f32_16x16x32_f16(false, a0, false, b0,
                                                       (short)0, c, false, false);
            c = __builtin_amdgcn_wmma_f32_16x16x32_f16(false, a1, false, b1,
                                                       (short)0, c, false, false);
            int u = nt * 16 + (lane & 15);
#pragma unroll
            for (int r = 0; r < 8; ++r) {
                int g = gg[r];
                if (g < 0) continue;
                float t = c[r];
                float* p = &s1[((size_t)g * 64 + u) * 3];
                atomicAdd(p + 0, t * sx[r]);
                atomicAdd(p + 1, t * sy[r]);
                atomicAdd(p + 2, t * sz[r]);
            }
        }
    }
}

// M2 = (L1o @ L_out) / 64, padded to 64 x 96 (cols >= 85 are zero)
__global__ void m2_kernel(const float* __restrict__ L1o,
                          const float* __restrict__ L_out,
                          float* __restrict__ M2) {
    int i = blockIdx.x * blockDim.x + threadIdx.x;
    if (i >= 64 * 96) return;
    int w = i / 96, v = i % 96;
    float acc = 0.f;
    if (v < MOUT)
        for (int t = 0; t < 64; ++t) acc += L1o[w * 64 + t] * L_out[t * MOUT + v];
    M2[i] = acc * (1.0f / 64.0f);
}

// out[g, v*3+m3] = (1 + 1/max(cnt,1)) * sum_w s1[g,w,m3] * M2[w,v]
__global__ __launch_bounds__(128) void out_kernel(
    const float* __restrict__ s1, const float* __restrict__ cnt,
    const float* __restrict__ M2, float* __restrict__ out) {
    __shared__ alignas(32) _Float16 bf[2 * 6 * 512];
    const int tid = threadIdx.x;
    for (int i = tid; i < 2 * 6 * 512; i += blockDim.x) {
        int e = i & 15, lane = (i >> 4) & 31, rest = i >> 9;
        int nt = rest % 6, kt = rest / 6;
        int k = kt * 32 + kmap(lane, e);
        int v = nt * 16 + (lane & 15);
        bf[i] = (_Float16)M2[k * 96 + v];
    }
    __syncthreads();

    const int wave = tid >> 5, lane = tid & 31;
    const int task = blockIdx.x * 4 + wave;          // 512 gtiles * 3 comps
    if (task >= (GG / 16) * 3) return;
    const int gtile = task / 3, m3 = task % 3;

    const int g = gtile * 16 + (lane & 15);
    const float scale = 1.0f + 1.0f / fmaxf(cnt[g], 1.0f);
    v16h a0, a1;
#pragma unroll
    for (int e = 0; e < 16; ++e) {
        int k = kmap(lane, e);
        a0[e] = (_Float16)(s1[((size_t)g * 64 + k) * 3 + m3] * scale);
        a1[e] = (_Float16)(s1[((size_t)g * 64 + 32 + k) * 3 + m3] * scale);
    }
    const int rowoff = (lane >> 4) * 8;
#pragma unroll
    for (int nt = 0; nt < 6; ++nt) {
        v8f c = {};
        const v16h b0 = *(const v16h*)&bf[((0 * 6 + nt) * 32 + lane) * 16];
        const v16h b1 = *(const v16h*)&bf[((1 * 6 + nt) * 32 + lane) * 16];
        c = __builtin_amdgcn_wmma_f32_16x16x32_f16(false, a0, false, b0,
                                                   (short)0, c, false, false);
        c = __builtin_amdgcn_wmma_f32_16x16x32_f16(false, a1, false, b1,
                                                   (short)0, c, false, false);
        int v = nt * 16 + (lane & 15);
        if (v < MOUT) {
#pragma unroll
            for (int r = 0; r < 8; ++r)
                out[(size_t)(gtile * 16 + rowoff + r) * (MOUT * 3) + v * 3 + m3] = c[r];
        }
    }
}

extern "C" void kernel_launch(void* const* d_in, const int* in_sizes, int n_in,
                              void* d_out, int out_size, void* d_ws, size_t ws_size,
                              hipStream_t stream) {
    const float* pos        = (const float*)d_in[0];
    const int*   z          = (const int*)  d_in[1];
    const int*   batch      = (const int*)  d_in[2];
    const float* node_table = (const float*)d_in[3];
    const float* W_rbf0 = (const float*)d_in[4];
    const float* b_rbf0 = (const float*)d_in[5];
    const float* W_rbf1 = (const float*)d_in[6];
    const float* b_rbf1 = (const float*)d_in[7];
    const float* W_m0   = (const float*)d_in[8];
    const float* b_m0   = (const float*)d_in[9];
    const float* W_m1   = (const float*)d_in[10];
    const float* b_m1   = (const float*)d_in[11];
    const float* W_m2   = (const float*)d_in[12];
    const float* b_m2   = (const float*)d_in[13];
    const float* w_tp1  = (const float*)d_in[15];
    const float* L1o    = (const float*)d_in[18];
    const float* L_out  = (const float*)d_in[20];

    float* cnt = (float*)d_ws;                    // G
    float* s1  = cnt + GG;                        // G*64*3
    float* M2  = s1 + (size_t)GG * 64 * 3;        // 64*96

    hipMemsetAsync(d_ws, 0, (size_t)(GG + (size_t)GG * 64 * 3) * sizeof(float), stream);
    m2_kernel<<<(64 * 96 + 255) / 256, 256, 0, stream>>>(L1o, L_out, M2);

    int tiles  = (NNODES + 15) / 16;
    int blocks = (tiles + WAVES - 1) / WAVES;
    node_kernel<<<blocks, 128, 0, stream>>>(pos, z, batch, node_table,
        W_rbf0, b_rbf0, W_rbf1, b_rbf1, W_m0, b_m0, W_m1, b_m1, W_m2, b_m2,
        w_tp1, cnt, s1, NNODES);

    int tasks = (GG / 16) * 3;
    out_kernel<<<(tasks + 3) / 4, 128, 0, stream>>>(s1, cnt, M2, (float*)d_out);
}
